// MNISTLabeledIndependantEncoder_7378753815094
// MI455X (gfx1250) — matrix operand order repaired and checked by reading, compiled, not run on hardware
//
#include <hip/hip_runtime.h>
#include <hip/hip_bf16.h>
#include <math.h>

// ---------------- problem constants ----------------
#define IN_DIM   784
#define KPAD     800      // IN_DIM padded to multiple of 32
#define HIDDEN   512
#define STYLE    128
#define BATCH    2048
#define NLAB     10

#define XS_STRIDE 808     // LDS stride for X tile (elements, 16B aligned)
#define H1_STRIDE 520     // LDS stride for h1 tile (elements, 16B aligned)

// ---------------- WMMA fragment types ----------------
typedef __attribute__((ext_vector_type(16))) __bf16        v16bf;
typedef __attribute__((ext_vector_type(8)))  float         v8f;
typedef __attribute__((ext_vector_type(4)))  unsigned int  v4u;

union Frag {
    v16bf bf;
    v4u   u[2];   // u[0] = elements 0..7, u[1] = elements 8..15
};

// float -> bf16 (round to nearest even); inputs are finite here
__device__ __forceinline__ unsigned short f2bf(float f) {
    unsigned int u = __float_as_uint(f);
    u += 0x7FFFu + ((u >> 16) & 1u);
    return (unsigned short)(u >> 16);
}

// ---------------- kernel 1: group rows by label (counting sort) ----------------
__global__ __launch_bounds__(256) void group_kernel(const int* __restrict__ m,
                                                    int* __restrict__ counts,
                                                    int* __restrict__ offsets,
                                                    int* __restrict__ perm,
                                                    int n) {
    __shared__ int cnt[16];
    __shared__ int cur[16];
    int t = threadIdx.x;
    if (t < 16) cnt[t] = 0;
    __syncthreads();
    for (int i = t; i < n; i += blockDim.x) atomicAdd(&cnt[m[i]], 1);
    __syncthreads();
    if (t == 0) {
        int acc = 0;
        for (int g = 0; g < NLAB; ++g) {
            offsets[g] = acc;
            cur[g]     = acc;
            counts[g]  = cnt[g];
            acc += cnt[g];
        }
    }
    __syncthreads();
    for (int i = t; i < n; i += blockDim.x) {
        int p = atomicAdd(&cur[m[i]], 1);
        perm[p] = i;
    }
}

// ---------------- kernel 2/3: transpose + bf16-convert weight tables ----------------
// w1t layout: [label][n(512)][k(KPAD)] ; k >= IN_DIM zero-padded, bias row excluded
__global__ __launch_bounds__(256) void convert_w1(const float* __restrict__ src,
                                                  unsigned short* __restrict__ dst) {
    int idx = blockIdx.x * 256 + threadIdx.x;
    const int total = NLAB * HIDDEN * KPAD;
    if (idx >= total) return;
    int g   = idx / (HIDDEN * KPAD);
    int rem = idx - g * (HIDDEN * KPAD);
    int n   = rem / KPAD;
    int k   = rem - n * KPAD;
    float v = (k < IN_DIM)
                ? src[(size_t)g * ((IN_DIM + 1) * HIDDEN) + (size_t)k * HIDDEN + n]
                : 0.0f;
    dst[idx] = f2bf(v);
}

// w2t layout: [label][n(128)][k(512)] ; bias row excluded
__global__ __launch_bounds__(256) void convert_w2(const float* __restrict__ src,
                                                  unsigned short* __restrict__ dst) {
    int idx = blockIdx.x * 256 + threadIdx.x;
    const int total = NLAB * STYLE * HIDDEN;
    if (idx >= total) return;
    int g   = idx / (STYLE * HIDDEN);
    int rem = idx - g * (STYLE * HIDDEN);
    int n   = rem / HIDDEN;
    int k   = rem - n * HIDDEN;
    dst[idx] = f2bf(src[(size_t)g * ((HIDDEN + 1) * STYLE) + (size_t)k * STYLE + n]);
}

// ---------------- kernel 4: fused grouped GEMM + relu + GEMM + sigmoid ----------------
// grid: (128 tiles, 10 labels) ; block: 256 threads = 8 wave32
__global__ __launch_bounds__(256) void encoder_kernel(
        const float* __restrict__ x,
        const float* __restrict__ fc1_tab,   // f32, for bias rows
        const float* __restrict__ fc2_tab,   // f32, for bias rows
        const int* __restrict__ counts,
        const int* __restrict__ offsets,
        const int* __restrict__ perm,
        const unsigned short* __restrict__ w1t,
        const unsigned short* __restrict__ w2t,
        float* __restrict__ out) {
    const int g    = blockIdx.y;
    const int tile = blockIdx.x;
    const int cnt  = counts[g];
    if (tile * 16 >= cnt) return;           // block-uniform exit
    const int base = offsets[g];

    __shared__ __align__(16) unsigned short xs[16 * XS_STRIDE];   // 25.3 KB
    __shared__ __align__(16) unsigned short h1s[16 * H1_STRIDE];  // 16.3 KB
    __shared__ int rowIdx[16];
    __shared__ int rowValid[16];

    const int t = threadIdx.x;
    if (t < 16) {
        int rr = tile * 16 + t;
        int cl = (rr < cnt) ? rr : (cnt - 1);  // clamp: duplicate last row, masked store
        rowIdx[t]   = perm[base + cl];
        rowValid[t] = (rr < cnt) ? 1 : 0;
    }
    __syncthreads();

    // ---- stage X tile into LDS as bf16, K padded to 800 ----
    for (int idx = t; idx < 16 * KPAD; idx += 256) {
        int r = idx / KPAD;
        int k = idx - r * KPAD;
        float v = (k < IN_DIM) ? x[(size_t)rowIdx[r] * IN_DIM + k] : 0.0f;
        xs[r * XS_STRIDE + k] = f2bf(v);
    }
    __syncthreads();

    const int wave = t >> 5;
    const int lane = t & 31;
    const int hi16 = lane >> 4;      // which 16-lane half
    const int l15  = lane & 15;

    // ================= FC1: (16 x 784) x (784 x 512) =================
    const unsigned short* w1g = w1t + (size_t)g * HIDDEN * KPAD;
    const int n_base = wave * 64;    // each wave owns 4 N-tiles of 16

    // per-lane base pointers; K-step becomes an immediate offset after unroll
    const unsigned short* apx = &xs[l15 * XS_STRIDE + hi16 * 16];
    const unsigned short* bp0 = w1g + (size_t)(n_base + 0 * 16 + l15) * KPAD + hi16 * 16;
    const unsigned short* bp1 = w1g + (size_t)(n_base + 1 * 16 + l15) * KPAD + hi16 * 16;
    const unsigned short* bp2 = w1g + (size_t)(n_base + 2 * 16 + l15) * KPAD + hi16 * 16;
    const unsigned short* bp3 = w1g + (size_t)(n_base + 3 * 16 + l15) * KPAD + hi16 * 16;

    v8f acc0 = v8f{}, acc1 = v8f{}, acc2 = v8f{}, acc3 = v8f{};

#pragma unroll
    for (int ks = 0; ks < KPAD / 32; ++ks) {
        const int ko = ks * 32;
        // A fragment (16-bit 16x32 layout): row = l15; two 8-elem chunks
        Frag a;
        a.u[0] = *(const v4u*)(apx + ko);
        a.u[1] = *(const v4u*)(apx + ko + 8);

        Frag b0, b1, b2, b3;
        b0.u[0] = *(const v4u*)(bp0 + ko);  b0.u[1] = *(const v4u*)(bp0 + ko + 8);
        b1.u[0] = *(const v4u*)(bp1 + ko);  b1.u[1] = *(const v4u*)(bp1 + ko + 8);
        b2.u[0] = *(const v4u*)(bp2 + ko);  b2.u[1] = *(const v4u*)(bp2 + ko + 8);
        b3.u[0] = *(const v4u*)(bp3 + ko);  b3.u[1] = *(const v4u*)(bp3 + ko + 8);

        acc0 = __builtin_amdgcn_wmma_f32_16x16x32_bf16(false, a.bf, false, b0.bf, (short)0, acc0, false, false);
        acc1 = __builtin_amdgcn_wmma_f32_16x16x32_bf16(false, a.bf, false, b1.bf, (short)0, acc1, false, false);
        acc2 = __builtin_amdgcn_wmma_f32_16x16x32_bf16(false, a.bf, false, b2.bf, (short)0, acc2, false, false);
        acc3 = __builtin_amdgcn_wmma_f32_16x16x32_bf16(false, a.bf, false, b3.bf, (short)0, acc3, false, false);
    }

    // bias + relu -> bf16 h1 in LDS (C/D layout: VGPR v -> M = v + 8*hi16, N = l15)
    const float* b1p = fc1_tab + (size_t)g * ((IN_DIM + 1) * HIDDEN) + (size_t)IN_DIM * HIDDEN;
    {
        const v8f* accs[4] = {&acc0, &acc1, &acc2, &acc3};
#pragma unroll
        for (int j = 0; j < 4; ++j) {
            int col = n_base + j * 16 + l15;
            float bias = b1p[col];
            const v8f& a8 = *accs[j];
#pragma unroll
            for (int v = 0; v < 8; ++v) {
                int row = v + hi16 * 8;
                float hv = a8[v] + bias;
                hv = hv > 0.0f ? hv : 0.0f;
                h1s[row * H1_STRIDE + col] = f2bf(hv);
            }
        }
    }
    __syncthreads();

    // ================= FC2: (16 x 512) x (512 x 128) =================
    const unsigned short* w2g = w2t + (size_t)g * STYLE * HIDDEN;
    const int n0 = wave * 16;        // one N-tile per wave
    const unsigned short* aph = &h1s[l15 * H1_STRIDE + hi16 * 16];
    const unsigned short* bph = w2g + (size_t)(n0 + l15) * HIDDEN + hi16 * 16;

    v8f acch = v8f{};

#pragma unroll
    for (int ks = 0; ks < HIDDEN / 32; ++ks) {
        const int ko = ks * 32;
        Frag a;
        a.u[0] = *(const v4u*)(aph + ko);
        a.u[1] = *(const v4u*)(aph + ko + 8);

        Frag b;
        b.u[0] = *(const v4u*)(bph + ko);
        b.u[1] = *(const v4u*)(bph + ko + 8);

        acch = __builtin_amdgcn_wmma_f32_16x16x32_bf16(false, a.bf, false, b.bf, (short)0, acch, false, false);
    }

    // bias + sigmoid, scatter valid rows
    const float* b2p = fc2_tab + (size_t)g * ((HIDDEN + 1) * STYLE) + (size_t)HIDDEN * STYLE;
    const int col = n0 + l15;
    const float bias2 = b2p[col];
#pragma unroll
    for (int v = 0; v < 8; ++v) {
        int row = v + hi16 * 8;
        if (rowValid[row]) {
            float hv = acch[v] + bias2;
            out[(size_t)rowIdx[row] * STYLE + col] = 1.0f / (1.0f + __expf(-hv));
        }
    }
}

// ---------------- host launcher ----------------
extern "C" void kernel_launch(void* const* d_in, const int* in_sizes, int n_in,
                              void* d_out, int out_size, void* d_ws, size_t ws_size,
                              hipStream_t stream) {
    (void)in_sizes; (void)n_in; (void)out_size; (void)ws_size;
    const float* x        = (const float*)d_in[0];
    const int*   m        = (const int*)d_in[1];
    const float* fc1_tab  = (const float*)d_in[2];
    const float* fc2_tab  = (const float*)d_in[3];
    float*       out      = (float*)d_out;

    char* ws = (char*)d_ws;
    int* counts  = (int*)(ws);          // 16 ints
    int* offsets = (int*)(ws + 64);     // 16 ints
    int* perm    = (int*)(ws + 128);    // BATCH ints -> ends at 128 + 8192 = 8320
    unsigned short* w1t = (unsigned short*)(ws + 8448);                     // 10*512*800*2 = 8,192,000 B
    unsigned short* w2t = (unsigned short*)(ws + 8448 + (size_t)NLAB * HIDDEN * KPAD * 2);
    // total workspace use: ~9.5 MB

    group_kernel<<<1, 256, 0, stream>>>(m, counts, offsets, perm, BATCH);

    {
        int total = NLAB * HIDDEN * KPAD;
        convert_w1<<<(total + 255) / 256, 256, 0, stream>>>(fc1_tab, w1t);
    }
    {
        int total = NLAB * STYLE * HIDDEN;
        convert_w2<<<(total + 255) / 256, 256, 0, stream>>>(fc2_tab, w2t);
    }

    dim3 grid(BATCH / 16, NLAB);   // worst case: all rows in one label
    encoder_kernel<<<grid, 256, 0, stream>>>(x, fc1_tab, fc2_tab,
                                             counts, offsets, perm,
                                             w1t, w2t, out);
}
